// BinConv2d_76493367542677
// MI455X (gfx1250) — 8-bit matrix operand order repaired and checked by reading, compiled, not run on hardware
//
#include <hip/hip_runtime.h>
#include <hip/hip_bf16.h>
#include <stdint.h>


#define CIN   128
#define KOUT  256
#define HW    56
#define OHW   (HW*HW)
#define NPIX  (32*OHW)
#define XN    (32*CIN*OHW)
#define WN    (KOUT*CIN*9)

typedef __attribute__((ext_vector_type(16))) int   i32x16;
typedef __attribute__((ext_vector_type(8)))  float f32x8;

#define ASYNC_LD_B128(ldsaddr, goff, base, OFF)                                 \
    asm volatile("global_load_async_to_lds_b128 %0, %1, %2 offset:" OFF         \
                 :: "v"(ldsaddr), "v"(goff), "s"(base) : "memory")
#define WAIT_ASYNC0() asm volatile("s_wait_asynccnt 0x0" ::: "memory")

__global__ void k_init(float* __restrict__ sums) {
    if (threadIdx.x == 0) { sums[0] = 0.0f; sums[1] = 0.0f; }
}

__global__ __launch_bounds__(256) void k_binx(const float* __restrict__ x,
                                              unsigned char* __restrict__ xq,
                                              float* __restrict__ sums) {
    __shared__ float red[256];
    float acc = 0.0f;
    const int stride = gridDim.x * blockDim.x;
    for (int idx = blockIdx.x * blockDim.x + threadIdx.x; idx < XN; idx += stride) {
        const int c  = idx & 127;
        const int hw = (idx >> 7) % OHW;
        const int n  = idx / (OHW * CIN);
        const float v = x[(n * CIN + c) * OHW + hw];
        xq[idx] = (v > 0.0f) ? 0x38 : ((v < 0.0f) ? 0xB8 : 0x00);
        acc += fabsf(v);
    }
    red[threadIdx.x] = acc;
    __syncthreads();
    for (int s = 128; s > 0; s >>= 1) {
        if ((int)threadIdx.x < s) red[threadIdx.x] += red[threadIdx.x + s];
        __syncthreads();
    }
    if (threadIdx.x == 0) atomicAdd(&sums[0], red[0]);
}

__global__ __launch_bounds__(256) void k_binw(const float* __restrict__ w,
                                              unsigned char* __restrict__ wq,
                                              float* __restrict__ sums) {
    __shared__ float red[256];
    float acc = 0.0f;
    const int stride = gridDim.x * blockDim.x;
    for (int idx = blockIdx.x * blockDim.x + threadIdx.x; idx < WN; idx += stride) {
        const float v = w[idx];
        const int k = idx / (CIN * 9);
        const int c = (idx / 9) % CIN;
        const int t = idx % 9;
        wq[(t * KOUT + k) * CIN + c] = (v > 0.0f) ? 0x38 : ((v < 0.0f) ? 0xB8 : 0x00);
        acc += fabsf(v);
    }
    red[threadIdx.x] = acc;
    __syncthreads();
    for (int s = 128; s > 0; s >>= 1) {
        if ((int)threadIdx.x < s) red[threadIdx.x] += red[threadIdx.x + s];
        __syncthreads();
    }
    if (threadIdx.x == 0) atomicAdd(&sums[1], red[0]);
}

#define LDT 144

__global__ __launch_bounds__(256) void k_conv(const unsigned char* __restrict__ xq,
                                              const unsigned char* __restrict__ wq,
                                              const float* __restrict__ sums,
                                              const float* __restrict__ bias,
                                              float* __restrict__ out) {
    __shared__ unsigned char ldsW[2][128 * LDT];
    __shared__ unsigned char ldsP[2][128 * LDT];

    const int tid   = threadIdx.x;
    const int lane  = tid & 31;
    const int wv    = tid >> 5;
    const int l15   = lane & 15;
    const int hi    = lane >> 4;
    const int waveM = (wv >> 1) * 32;
    const int waveN = (wv & 1) * 64;
    const int pixBase = blockIdx.x * 128;
    const int kBase   = blockIdx.y * 128;

    f32x8 acc[2][4];
#pragma unroll
    for (int mi = 0; mi < 2; ++mi)
#pragma unroll
        for (int ni = 0; ni < 4; ++ni) acc[mi][ni] = (f32x8){};

    const int r    = tid >> 1;
    const int half = tid & 1;
    const int p    = pixBase + r;
    const int pn   = p / OHW;
    const int phw  = p % OHW;
    const int poh  = phw / HW;
    const int pow_ = phw % HW;

    auto fill = [&](int buf, int tap) {
        const int kh = tap / 3, kw = tap % 3;
        {
            const unsigned int lds  = (unsigned int)(size_t)&ldsW[buf][r * LDT + half * 64];
            const unsigned int goff = (unsigned int)((tap * KOUT + kBase + r) * CIN + half * 64);
            ASYNC_LD_B128(lds, goff, wq, "0");
            ASYNC_LD_B128(lds, goff, wq, "16");
            ASYNC_LD_B128(lds, goff, wq, "32");
            ASYNC_LD_B128(lds, goff, wq, "48");
        }
        {
            const int ih = poh + kh - 1;
            const int iw = pow_ + kw - 1;
            const unsigned int lds = (unsigned int)(size_t)&ldsP[buf][r * LDT + half * 64];
            if (ih >= 0 && ih < HW && iw >= 0 && iw < HW) {
                const unsigned int goff =
                    (unsigned int)(((pn * HW + ih) * HW + iw) * CIN + half * 64);
                ASYNC_LD_B128(lds, goff, xq, "0");
                ASYNC_LD_B128(lds, goff, xq, "16");
                ASYNC_LD_B128(lds, goff, xq, "32");
                ASYNC_LD_B128(lds, goff, xq, "48");
            } else {
                uint4 z; z.x = 0; z.y = 0; z.z = 0; z.w = 0;
                uint4* dst = (uint4*)&ldsP[buf][r * LDT + half * 64];
#pragma unroll
                for (int i = 0; i < 4; ++i) dst[i] = z;
            }
        }
    };

    fill(0, 0);
    WAIT_ASYNC0();
    __syncthreads();

    for (int tap = 0; tap < 9; ++tap) {
        const int cur = tap & 1;
        if (tap < 8) fill(cur ^ 1, tap + 1);

        i32x16 bfrag[4];
#pragma unroll
        for (int ni = 0; ni < 4; ++ni) {
            const unsigned char* base = &ldsP[cur][(waveN + ni * 16 + l15) * LDT + hi * 16];
#pragma unroll
            for (int q = 0; q < 4; ++q) {
                const uint4 d = *(const uint4*)(base + q * 32);
                bfrag[ni][q * 4 + 0] = (int)d.x;
                bfrag[ni][q * 4 + 1] = (int)d.y;
                bfrag[ni][q * 4 + 2] = (int)d.z;
                bfrag[ni][q * 4 + 3] = (int)d.w;
            }
        }

#pragma unroll
        for (int mi = 0; mi < 2; ++mi) {
            i32x16 afrag;
            const unsigned char* base = &ldsW[cur][(waveM + mi * 16 + l15) * LDT + hi * 16];
#pragma unroll
            for (int h = 0; h < 2; ++h)
#pragma unroll
                for (int g = 0; g < 4; ++g) {
                    const uint2 d = *(const uint2*)(base + h * 64 + (g >> 1) * 32 + (g & 1) * 8);
                    afrag[h * 8 + g * 2 + 0] = (int)d.x;
                    afrag[h * 8 + g * 2 + 1] = (int)d.y;
                }
#pragma unroll
            for (int ni = 0; ni < 4; ++ni)
                acc[mi][ni] = __builtin_amdgcn_wmma_f32_16x16x128_fp8_fp8(
                    afrag, bfrag[ni], (short)0, acc[mi][ni],
                     false,  false);
        }

        WAIT_ASYNC0();
        __syncthreads();
    }

    const float scale = (sums[0] * (1.0f / (float)XN)) * (sums[1] * (1.0f / (float)WN));
#pragma unroll
    for (int ni = 0; ni < 4; ++ni) {
        const int pcol = pixBase + waveN + ni * 16 + l15;
        const int n    = pcol / OHW;
        const int hw   = pcol % OHW;
        const int obase = n * (KOUT * OHW) + hw;
#pragma unroll
        for (int mi = 0; mi < 2; ++mi)
#pragma unroll
            for (int j = 0; j < 8; ++j) {
                const int k = kBase + waveM + mi * 16 + j + 8 * hi;
                out[obase + k * OHW] = acc[mi][ni][j] * scale + bias[k];
            }
    }
}

extern "C" void kernel_launch(void* const* d_in, const int* in_sizes, int n_in,
                              void* d_out, int out_size, void* d_ws, size_t ws_size,
                              hipStream_t stream) {
    const float* x    = (const float*)d_in[0];
    const float* w    = (const float*)d_in[1];
    const float* bias = (const float*)d_in[2];
    float* out = (float*)d_out;

    float* sums       = (float*)d_ws;
    unsigned char* xq = (unsigned char*)d_ws + 256;
    unsigned char* wq = xq + (size_t)XN;

    k_init<<<1, 64, 0, stream>>>(sums);
    k_binx<<<2048, 256, 0, stream>>>(x, xq, sums);
    k_binw<<<144, 256, 0, stream>>>(w, wq, sums);

    dim3 grid(NPIX / 128, KOUT / 128);
    k_conv<<<grid, 256, 0, stream>>>(xq, wq, sums, bias, out);
}
